// MADGCNN_54760833024631
// MI455X (gfx1250) — matrix operand order repaired and checked by reading, not compiled, not run on hardware
//
#include <hip/hip_runtime.h>
#include <hip/hip_bf16.h>

typedef _Float16 h16;
typedef __attribute__((ext_vector_type(16))) _Float16 v16h;
typedef __attribute__((ext_vector_type(8)))  _Float16 v8h;
typedef __attribute__((ext_vector_type(8)))  float    v8f;

union V16U { v16h v; v8h h[2]; };

// ---------------------------------------------------------------------------
// Wave-level GEMM strip: NT adjacent 16x16 output tiles sharing one A fragment.
//   D[m, p+j*16] += sum_k W[mbase+m, k] * Act[pbase+j*16+p, k]
// Kd % 32 == 0. W row-major (ldw), Act row-major (lda).
// A fragment loaded once per K-step and reused across NT B fragments ->
// NT independent WMMA accumulation chains (hides XDL latency) + 1/NT A traffic.
// k order used here (contiguous, same for A and B): A lanes 0-15 rows M=0..15 K=0..15, lanes 16-31
// same rows K=16..31; B lanes 0-15 col=lane K=0..15, lanes 16-31 K=16..31.
// ---------------------------------------------------------------------------
template<int NT>
__device__ __forceinline__ void wmma_strip(const h16* __restrict__ Wp, int ldw, int mbase,
                                           const h16* __restrict__ Ap, int lda, int pbase,
                                           int Kd, int lane, v8f* acc)
{
  const int hi = lane >> 4;
  const h16* a = Wp + (size_t)(mbase + (lane & 15)) * ldw + 16 * hi;
  const h16* b = Ap + (size_t)(pbase + (lane & 15)) * lda + 16 * hi;
  for (int kk = 0; kk < Kd; kk += 32) {
    V16U av;
    av.h[0] = *(const v8h*)(a + kk);
    av.h[1] = *(const v8h*)(a + kk + 8);
#pragma unroll
    for (int j = 0; j < NT; ++j) {
      v16h bv = *(const v16h*)(b + (size_t)j * 16 * lda + kk);
      acc[j] = __builtin_amdgcn_wmma_f32_16x16x32_f16(false, av.v, false, bv,
                                                      (short)0, acc[j], false, false);
    }
  }
}

// ---------------------------------------------------------------------------
// Prep kernels
// ---------------------------------------------------------------------------
__global__ void k_f32_to_f16(const float* __restrict__ s, h16* __restrict__ d, int n)
{
  int i = blockIdx.x * blockDim.x + threadIdx.x;
  if (i < n) d[i] = (h16)s[i];
}

__global__ void k_pad_w1(const float* __restrict__ w1, h16* __restrict__ d)
{
  int t = blockIdx.x * blockDim.x + threadIdx.x;   // 64*32
  if (t >= 64 * 32) return;
  int o = t >> 5, k = t & 31;
  d[t] = (h16)((k < 6) ? w1[o * 6 + k] : 0.0f);
}

__global__ void k_fold_bn(const float* __restrict__ g, const float* __restrict__ b,
                          const float* __restrict__ m, const float* __restrict__ v,
                          float* __restrict__ sc, float* __restrict__ bi, int n)
{
  int i = blockIdx.x * blockDim.x + threadIdx.x;
  if (i >= n) return;
  float s = g[i] * rsqrtf(v[i] + 1e-5f);
  sc[i] = s;
  bi[i] = b[i] - m[i] * s;
}

__global__ void k_prep_x(const float* __restrict__ x, h16* __restrict__ act0)
{
  int t = blockIdx.x * blockDim.x + threadIdx.x;   // 8192*32
  if (t >= 8192 * 32) return;
  int pos = t >> 5, k = t & 31;
  int b = pos >> 10, n = pos & 1023;
  float v = (k < 3) ? x[(b * 3 + k) * 1024 + n] : 0.0f;
  act0[t] = (h16)v;
}

// ---------------------------------------------------------------------------
// kNN: squared norms, Gram matrix (WMMA), top-k
// ---------------------------------------------------------------------------
__global__ void k_sqnorm(const h16* __restrict__ act, int lda, int Kd, float* __restrict__ sq)
{
  int pos = blockIdx.x * blockDim.x + threadIdx.x;
  if (pos >= 8192) return;
  const h16* p = act + (size_t)pos * lda;
  float s = 0.f;
  for (int k = 0; k < Kd; ++k) { float v = (float)p[k]; s += v * v; }
  sq[pos] = s;
}

// Block covers a 128(M) x 64(P) region; each of 8 waves owns one M-tile and
// 4 P-tiles (4 independent accumulators).
__global__ void __launch_bounds__(256)
k_gram(const h16* __restrict__ act, int lda, int Kd,
       const float* __restrict__ sq, float* __restrict__ pd)
{
  const int N = 1024;
  int blk = blockIdx.x;                 // b*128 + rm*16 + rp
  int b  = blk >> 7;
  int rm = (blk >> 4) & 7, rp = blk & 15;
  int wave = threadIdx.x >> 5, lane = threadIdx.x & 31;
  const h16*   ab = act + (size_t)b * N * lda;
  const float* sb = sq + b * N;
  float*      pb_ = pd + (size_t)b * N * N;
  int mb = rm * 128 + wave * 16;
  int pb = rp * 64;
  v8f acc[4] = {{}, {}, {}, {}};
  wmma_strip<4>(ab, lda, mb, ab, lda, pb, Kd, lane, acc);
  int m0 = mb + 8 * (lane >> 4);
#pragma unroll
  for (int j = 0; j < 4; ++j) {
    int p = pb + j * 16 + (lane & 15);
    float sp = sb[p];
#pragma unroll
    for (int v = 0; v < 8; ++v) {
      int m = m0 + v;
      pb_[(size_t)m * N + p] = 2.0f * acc[j][v] - sb[m] - sp;
    }
  }
}

__global__ void k_topk(const float* __restrict__ pd, int* __restrict__ idx)
{
  const int N = 1024, KNN = 20;
  int pos = blockIdx.x * blockDim.x + threadIdx.x;
  if (pos >= 8192) return;
  const float* row = pd + (size_t)pos * N;
  float bv[KNN]; int bj[KNN];
#pragma unroll
  for (int i = 0; i < KNN; ++i) { bv[i] = -1e30f; bj[i] = 0; }
  for (int j = 0; j < N; ++j) {
    float v = row[j];
    if (v > bv[KNN - 1]) {
      int t = KNN - 1;
      while (t > 0 && bv[t - 1] < v) { bv[t] = bv[t - 1]; bj[t] = bj[t - 1]; --t; }
      bv[t] = v; bj[t] = j;
    }
  }
  for (int i = 0; i < KNN; ++i) idx[pos * KNN + i] = bj[i];
}

// ---------------------------------------------------------------------------
// Fused edge conv: gather edge features -> LDS, WMMA conv, BN+LReLU, max over k
// One block per point. feat: 32 cols (20 valid) x Kpad f16; hout: O x 32 f32.
// Each wave owns one M-tile across both P-tiles (2 accumulators, shared A).
// ---------------------------------------------------------------------------
__global__ void __launch_bounds__(256)
k_edge_conv(const h16* __restrict__ act, int lda, int Cin, int Kpad,
            const int* __restrict__ idx, const h16* __restrict__ Wp, int O,
            const float* __restrict__ sc, const float* __restrict__ bi,
            h16* __restrict__ outp)
{
  const int N = 1024, KNN = 20;
  __shared__ h16  feat[32 * 256];
  __shared__ float hout[256 * 32];
  int pos = blockIdx.x;
  int b   = pos >> 10;
  int tid = threadIdx.x;
  const h16* ctr = act + (size_t)pos * lda;
  int total = 32 * Kpad;
  for (int e = tid; e < total; e += 256) {
    int col = e / Kpad, k = e - col * Kpad;
    h16 v = (h16)0.0f;
    if (col < KNN) {
      int nb = idx[pos * KNN + col];
      const h16* nbr = act + (size_t)(b * N + nb) * lda;
      if (k < Cin)          v = (h16)((float)nbr[k] - (float)ctr[k]);
      else if (k < 2 * Cin) v = ctr[k - Cin];
    }
    feat[col * Kpad + k] = v;
  }
  __syncthreads();
  int wave = tid >> 5, lane = tid & 31;
  int jobs = O >> 4;                           // one M-tile per job, both P-tiles
  for (int j = wave; j < jobs; j += 8) {
    v8f acc[2] = {{}, {}};
    wmma_strip<2>(Wp, Kpad, j * 16, feat, Kpad, 0, Kpad, lane, acc);
    int m0 = j * 16 + 8 * (lane >> 4);
#pragma unroll
    for (int jt = 0; jt < 2; ++jt) {
      int p = jt * 16 + (lane & 15);
#pragma unroll
      for (int v = 0; v < 8; ++v) {
        int m = m0 + v;
        float t = acc[jt][v] * sc[m] + bi[m];
        hout[m * 32 + p] = (t > 0.f) ? t : 0.2f * t;
      }
    }
  }
  __syncthreads();
  if (tid < O) {
    float mx = hout[tid * 32];
    for (int c = 1; c < KNN; ++c) mx = fmaxf(mx, hout[tid * 32 + c]);
    outp[(size_t)pos * O + tid] = (h16)mx;
  }
}

// ---------------------------------------------------------------------------
// Generic WMMA GEMM: out[p, m] = epilogue( sum_k W[m,k]*Act[p,k] )
// One wave computes a 16(M) x 64(P) strip (4 accumulators).
//   mode 0: f32 store, +bias[m]               (QKV projection)
//   mode 1: f16 store, +bias[m]+resid[p,m]    (attn out-proj + residual)
//   mode 2: f32 store, BN fold + LeakyReLU    (W5 embedding)
// ---------------------------------------------------------------------------
__global__ void __launch_bounds__(256)
k_gemm(const h16* __restrict__ Wp, int ldw,
       const h16* __restrict__ Ap, int lda,
       int nMt, int nPg, int Kd, int mode,
       const float* __restrict__ bias,
       const float* __restrict__ sc, const float* __restrict__ bi,
       const h16* __restrict__ resid, int ldres,
       float* __restrict__ outf, h16* __restrict__ outh, int ldo)
{
  int wid = blockIdx.x * 8 + (threadIdx.x >> 5);
  if (wid >= nMt * nPg) return;
  int lane = threadIdx.x & 31;
  int mt = wid % nMt, pg = wid / nMt;
  v8f acc[4] = {{}, {}, {}, {}};
  wmma_strip<4>(Wp, ldw, mt * 16, Ap, lda, pg * 64, Kd, lane, acc);
  int m0 = mt * 16 + 8 * (lane >> 4);
#pragma unroll
  for (int j = 0; j < 4; ++j) {
    int p = pg * 64 + j * 16 + (lane & 15);
#pragma unroll
    for (int v = 0; v < 8; ++v) {
      int m = m0 + v;
      float t = acc[j][v];
      if (mode == 0) {
        outf[(size_t)p * ldo + m] = t + bias[m];
      } else if (mode == 1) {
        t += bias[m] + (float)resid[(size_t)p * ldres + m];
        outh[(size_t)p * ldo + m] = (h16)t;
      } else {
        t = t * sc[m] + bi[m];
        outf[(size_t)p * ldo + m] = (t > 0.f) ? t : 0.2f * t;
      }
    }
  }
}

// ---------------------------------------------------------------------------
// Attention core: seq len L=B=8, batch = N positions, per (n, head, l) thread.
// qkv f32 layout (pos=l*N+n, 3E). Writes o (pos, E) f16.
// ---------------------------------------------------------------------------
__global__ void k_attn(const float* __restrict__ qkv, int E, h16* __restrict__ obuf)
{
  const int N = 1024, L = 8;
  int tid = blockIdx.x * blockDim.x + threadIdx.x;
  if (tid >= N * 8 * 8) return;
  int n = tid >> 6;
  int h = (tid >> 3) & 7;
  int l = tid & 7;
  int Dh = E >> 3;
  int E3 = 3 * E;
  const float* qp = qkv + (size_t)(l * N + n) * E3 + h * Dh;
  float s[8], mx = -1e30f;
  for (int m = 0; m < L; ++m) {
    const float* kp = qkv + (size_t)(m * N + n) * E3 + E + h * Dh;
    float d = 0.f;
    for (int t = 0; t < Dh; ++t) d += qp[t] * kp[t];
    s[m] = d * rsqrtf((float)Dh);
    mx = fmaxf(mx, s[m]);
  }
  float sum = 0.f;
  for (int m = 0; m < L; ++m) { s[m] = __expf(s[m] - mx); sum += s[m]; }
  float inv = 1.0f / sum;
  h16* op = obuf + (size_t)(l * N + n) * E + h * Dh;
  for (int t = 0; t < Dh; ++t) {
    float o = 0.f;
    for (int m = 0; m < L; ++m)
      o += s[m] * qkv[(size_t)(m * N + n) * E3 + 2 * E + h * Dh + t];
    op[t] = (h16)(o * inv);
  }
}

// ---------------------------------------------------------------------------
// Global pooling: max + mean over N per (b, c)
// ---------------------------------------------------------------------------
__global__ void k_pool(const float* __restrict__ h5, float* __restrict__ g)
{
  const int N = 1024, EMB = 1024;
  int tid = blockIdx.x * blockDim.x + threadIdx.x;
  if (tid >= 8 * EMB) return;
  int b = tid >> 10, c = tid & 1023;
  const float* p = h5 + (size_t)(b * N) * EMB + c;
  float mx = -1e30f, sm = 0.f;
  for (int n = 0; n < N; ++n) { float v = p[(size_t)n * EMB]; mx = fmaxf(mx, v); sm += v; }
  g[b * 2 * EMB + c]       = mx;
  g[b * 2 * EMB + EMB + c] = sm * (1.0f / N);
}

// ---------------------------------------------------------------------------
// MLP head (B=8 rows): L1 -> bn6 -> lrelu -> L2 -> bn7 -> lrelu -> L3
// ---------------------------------------------------------------------------
__global__ void __launch_bounds__(256)
k_head(const float* __restrict__ g,
       const float* __restrict__ L1w,
       const float* g6g, const float* g6b, const float* g6m, const float* g6v,
       const float* __restrict__ L2w, const float* __restrict__ L2b,
       const float* g7g, const float* g7b, const float* g7m, const float* g7v,
       const float* __restrict__ L3w, const float* __restrict__ L3b,
       float* __restrict__ out)
{
  __shared__ float t0[2048];
  __shared__ float t1[512];
  __shared__ float t2[256];
  int b = blockIdx.x, tid = threadIdx.x;
  for (int i = tid; i < 2048; i += 256) t0[i] = g[b * 2048 + i];
  __syncthreads();
  for (int j = tid; j < 512; j += 256) {
    const float* w = L1w + (size_t)j * 2048;
    float d = 0.f;
    for (int i = 0; i < 2048; ++i) d += w[i] * t0[i];
    float t = (d - g6m[j]) * rsqrtf(g6v[j] + 1e-5f) * g6g[j] + g6b[j];
    t1[j] = (t > 0.f) ? t : 0.2f * t;
  }
  __syncthreads();
  if (tid < 256) {
    const float* w = L2w + (size_t)tid * 512;
    float d = 0.f;
    for (int i = 0; i < 512; ++i) d += w[i] * t1[i];
    d += L2b[tid];
    float t = (d - g7m[tid]) * rsqrtf(g7v[tid] + 1e-5f) * g7g[tid] + g7b[tid];
    t2[tid] = (t > 0.f) ? t : 0.2f * t;
  }
  __syncthreads();
  if (tid < 40) {
    const float* w = L3w + (size_t)tid * 256;
    float d = 0.f;
    for (int i = 0; i < 256; ++i) d += w[i] * t2[i];
    out[b * 40 + tid] = d + L3b[tid];
  }
}

// ---------------------------------------------------------------------------
// Orchestration
// ---------------------------------------------------------------------------
extern "C" void kernel_launch(void* const* d_in, const int* in_sizes, int n_in,
                              void* d_out, int out_size, void* d_ws, size_t ws_size,
                              hipStream_t stream)
{
  (void)in_sizes; (void)n_in; (void)out_size; (void)ws_size;
  const int Bn = 8, N = 1024, P = Bn * N, KNN = 20, EMB = 1024;

  size_t off = 0;
  auto alloc = [&](size_t bytes) -> char* {
    char* p = (char*)d_ws + off;
    off = (off + bytes + 255) & ~(size_t)255;
    return p;
  };
  h16*   act0 = (h16*)  alloc((size_t)P * 32 * 2);
  h16*   xcat = (h16*)  alloc((size_t)P * 512 * 2);
  h16*   tmp  = (h16*)  alloc((size_t)P * 256 * 2);
  float* qkv  = (float*)alloc((size_t)P * 768 * 4);
  h16*   obuf = (h16*)  alloc((size_t)P * 256 * 2);
  float* pd   = (float*)alloc((size_t)Bn * N * N * 4);
  float* sq   = (float*)alloc((size_t)P * 4);
  int*   idx  = (int*)  alloc((size_t)P * KNN * 4);
  float* h5   = (float*)alloc((size_t)P * EMB * 4);
  float* gb   = (float*)alloc((size_t)Bn * 2 * EMB * 4);

  int Cin[4] = {3, 64, 64, 128};
  int Kp[4]  = {32, 128, 128, 256};
  int Oc[4]  = {64, 64, 128, 256};
  int coff[4]= {0, 64, 128, 256};
  h16 *wE[4], *wIn[4], *wOut[4]; float *scE[4], *biE[4];
  for (int s = 0; s < 4; ++s) {
    wE[s]   = (h16*)  alloc((size_t)Oc[s] * Kp[s] * 2);
    wIn[s]  = (h16*)  alloc((size_t)3 * Oc[s] * Oc[s] * 2);
    wOut[s] = (h16*)  alloc((size_t)Oc[s] * Oc[s] * 2);
    scE[s]  = (float*)alloc(Oc[s] * 4);
    biE[s]  = (float*)alloc(Oc[s] * 4);
  }
  h16*   w5  = (h16*)  alloc((size_t)EMB * 512 * 2);
  float* sc5 = (float*)alloc(EMB * 4);
  float* bi5 = (float*)alloc(EMB * 4);

  // Input index map (setup_inputs insertion order, nested dicts flattened in order)
  int iW[4]    = {1, 10, 19, 28};
  int ibn[4]   = {2, 11, 20, 29};    // g,b,m,v consecutive
  int iinw[4]  = {6, 15, 24, 33};
  int iinb[4]  = {7, 16, 25, 34};
  int ioutw[4] = {8, 17, 26, 35};
  int ioutb[4] = {9, 18, 27, 36};

  // ---- prep ----
  k_pad_w1<<<8, 256, 0, stream>>>((const float*)d_in[1], wE[0]);
  for (int s = 1; s < 4; ++s) {
    int n = Oc[s] * Kp[s];
    k_f32_to_f16<<<(n + 255) / 256, 256, 0, stream>>>((const float*)d_in[iW[s]], wE[s], n);
  }
  for (int s = 0; s < 4; ++s) {
    int n = 3 * Oc[s] * Oc[s];
    k_f32_to_f16<<<(n + 255) / 256, 256, 0, stream>>>((const float*)d_in[iinw[s]], wIn[s], n);
    n = Oc[s] * Oc[s];
    k_f32_to_f16<<<(n + 255) / 256, 256, 0, stream>>>((const float*)d_in[ioutw[s]], wOut[s], n);
    k_fold_bn<<<(Oc[s] + 255) / 256, 256, 0, stream>>>(
        (const float*)d_in[ibn[s]], (const float*)d_in[ibn[s] + 1],
        (const float*)d_in[ibn[s] + 2], (const float*)d_in[ibn[s] + 3],
        scE[s], biE[s], Oc[s]);
  }
  { int n = EMB * 512;
    k_f32_to_f16<<<(n + 255) / 256, 256, 0, stream>>>((const float*)d_in[37], w5, n); }
  k_fold_bn<<<(EMB + 255) / 256, 256, 0, stream>>>(
      (const float*)d_in[38], (const float*)d_in[39],
      (const float*)d_in[40], (const float*)d_in[41], sc5, bi5, EMB);
  k_prep_x<<<(P * 32 + 255) / 256, 256, 0, stream>>>((const float*)d_in[0], act0);

  // ---- 4 stages: knn -> edge conv -> MHA ----
  const h16* actIn = act0;
  int lda = 32, KdG = 32;
  for (int s = 0; s < 4; ++s) {
    int E = Oc[s];
    k_sqnorm<<<(P + 255) / 256, 256, 0, stream>>>(actIn, lda, KdG, sq);
    k_gram<<<Bn * 128, 256, 0, stream>>>(actIn, lda, KdG, sq, pd);
    k_topk<<<(P + 255) / 256, 256, 0, stream>>>(pd, idx);
    k_edge_conv<<<P, 256, 0, stream>>>(actIn, lda, Cin[s], Kp[s], idx, wE[s], Oc[s],
                                       scE[s], biE[s], tmp);
    int nPg = (P / 16) / 4;   // 64-wide P strips
    int nMt = (3 * E) / 16;   // QKV projection
    k_gemm<<<(nMt * nPg + 7) / 8, 256, 0, stream>>>(
        wIn[s], E, tmp, E, nMt, nPg, E, 0,
        (const float*)d_in[iinb[s]], nullptr, nullptr, nullptr, 0,
        qkv, nullptr, 3 * E);
    k_attn<<<(N * 8 * 8 + 255) / 256, 256, 0, stream>>>(qkv, E, obuf);
    nMt = E / 16;             // out-proj + bias + residual -> xcat slice
    k_gemm<<<(nMt * nPg + 7) / 8, 256, 0, stream>>>(
        wOut[s], E, obuf, E, nMt, nPg, E, 1,
        (const float*)d_in[ioutb[s]], nullptr, nullptr, tmp, E,
        nullptr, xcat + coff[s], 512);
    actIn = xcat + coff[s]; lda = 512; KdG = E;
  }

  // ---- W5 embedding (K=512) with fused BN + LReLU ----
  { int nMt = EMB / 16, nPg = (P / 16) / 4;
    k_gemm<<<(nMt * nPg + 7) / 8, 256, 0, stream>>>(
        w5, 512, xcat, 512, nMt, nPg, 512, 2,
        nullptr, sc5, bi5, nullptr, 0, h5, nullptr, EMB);
  }
  k_pool<<<(Bn * EMB + 255) / 256, 256, 0, stream>>>(h5, gb);
  k_head<<<Bn, 256, 0, stream>>>(gb, (const float*)d_in[42],
      (const float*)d_in[43], (const float*)d_in[44], (const float*)d_in[45], (const float*)d_in[46],
      (const float*)d_in[47], (const float*)d_in[48],
      (const float*)d_in[49], (const float*)d_in[50], (const float*)d_in[51], (const float*)d_in[52],
      (const float*)d_in[53], (const float*)d_in[54],
      (float*)d_out);
}
